// FSInstSetCriterion_22883585753395
// MI455X (gfx1250) — compile-verified, matrix operand order checked by reading, not run on hardware
//
#include <hip/hip_runtime.h>
#include <math.h>

typedef __attribute__((ext_vector_type(16))) _Float16 v16h;
typedef __attribute__((ext_vector_type(8)))  float    v8f;

#define Bsz 4
#define Qn  256
#define Nn  40000
#define In  32
#define Gn  16
#define KSLICES 10
#define NSLICE  (Nn / KSLICES)   /* 4000, multiple of 32 */

__device__ __forceinline__ float bce_logits(float s, float t) {
  // numerically stable BCE-with-logits
  return fmaxf(s, 0.f) - s * t + log1pf(expf(-fabsf(s)));
}

__device__ __forceinline__ float blockReduce256(float v, volatile float* red, int tid) {
  red[tid] = v;
  __syncthreads();
  #pragma unroll
  for (int s = 128; s > 0; s >>= 1) {
    if (tid < s) red[tid] += red[tid + s];
    __syncthreads();
  }
  float r = red[0];
  __syncthreads();
  return r;
}

// -----------------------------------------------------------------------------
// Kernel 1: counts[b,q,i] = #{ n : mask_logits[b,q,n] > 0 && labels[b,n] == i }
// computed as a GEMM  (m f16 16x32) x (onehot f16 32x16) with f32 accumulate.
// One wave32 per block; two 16x16 WMMA accumulators cover I=32 (A reused).
// All labels for the slice are staged in LDS ONCE (16 KB), so the 125-step
// inner loop has no barriers and no scalar global label traffic.
// Grid: (qtile=16, b=4, kslice=KSLICES). Exact integer arithmetic throughout.
// -----------------------------------------------------------------------------
__global__ __launch_bounds__(32)
void counts_wmma_kernel(const float* __restrict__ logits,
                        const int*   __restrict__ labels,
                        float*       __restrict__ partial) {
  const int qt    = blockIdx.x;        // 0..15
  const int b     = blockIdx.y;        // 0..3
  const int slice = blockIdx.z;        // 0..KSLICES-1
  const int lane  = threadIdx.x;       // wave32
  const int q0    = qt * 16;
  const int mrow  = q0 + (lane & 15);
  const int kbase = (lane & 16) ? 8 : 0;  // A/B K pattern per ISA layout
  const int col   = lane & 15;
  const float* Abase = logits + (size_t)(b * Qn + mrow) * Nn + slice * NSLICE;
  const int*   Lrow  = labels + (size_t)b * Nn + slice * NSLICE;

  // Stage the whole slice's labels in LDS once (4000 ints = 16 KB).
  __shared__ int lsh[NSLICE];
  for (int i = lane; i < NSLICE; i += 32) lsh[i] = Lrow[i];
  __syncthreads();

  v8f acc0 = {};   // columns i = 0..15
  v8f acc1 = {};   // columns i = 16..31

  // half-wave-dependent prefetch offset: lanes 0-15 and 16-31 cover different
  // 256B regions of the same row, 4KB ahead of the current k-step.
  const int pf = 1024 + ((lane >> 4) << 6);

  for (int n = 0; n < NSLICE; n += 32) {
    __builtin_prefetch(Abase + n + pf, 0, 1);   // gfx1250 global_prefetch_b8

    // A tile 16x32 f16: lane holds M=lane&15, K = kbase+{0..7} and kbase+16+{0..7}
    union { float4 v4[4]; float f[16]; } av;
    av.v4[0] = *(const float4*)(Abase + n + kbase);
    av.v4[1] = *(const float4*)(Abase + n + kbase + 4);
    av.v4[2] = *(const float4*)(Abase + n + kbase + 16);
    av.v4[3] = *(const float4*)(Abase + n + kbase + 20);

    v16h a, b0, b1;
    #pragma unroll
    for (int j = 0; j < 16; ++j) {
      a[j] = (av.f[j] > 0.f) ? (_Float16)1.0f : (_Float16)0.0f;
      const int k  = n + kbase + ((j < 8) ? j : (8 + j));  // = +j  /  +16+(j-8)
      const int lv = lsh[k];
      b0[j] = (lv == col)        ? (_Float16)1.0f : (_Float16)0.0f;
      b1[j] = (lv == (col + 16)) ? (_Float16)1.0f : (_Float16)0.0f;
    }

    acc0 = __builtin_amdgcn_wmma_f32_16x16x32_f16(false, a, false, b0, (short)0, acc0, false, false);
    acc1 = __builtin_amdgcn_wmma_f32_16x16x32_f16(false, a, false, b1, (short)0, acc1, false, false);
  }

  // C/D layout: VGPR r -> M = r + (lane>=16 ? 8 : 0), N = lane&15
  const int rofs = (lane & 16) ? 8 : 0;
  float* base = partial + (size_t)((slice * Bsz + b) * Qn + q0) * In;
  #pragma unroll
  for (int r = 0; r < 8; ++r) {
    float* dst = base + (size_t)(r + rofs) * In;
    dst[col]      = acc0[r];
    dst[col + 16] = acc1[r];
  }
}

// -----------------------------------------------------------------------------
// Kernel 2: per-batch similarity loss partials.
//   inter = max(counts); |m| = sum(counts); union = |m| + hist[mode] - inter
//   then BCE, pos/neg classification, descending bitonic sort of loss_neg,
//   hard-negative selection.  Writes {num_pos, sum_loss_pos, sum_neg_sel} per b.
// -----------------------------------------------------------------------------
__global__ __launch_bounds__(256)
void sim_kernel(const float* __restrict__ sim_score,
                const int*   __restrict__ labels,
                const float* __restrict__ partial,
                float*       __restrict__ simpart) {
  const int b   = blockIdx.x;
  const int tid = threadIdx.x;   // one query per thread (Q = 256)
  __shared__ int   hist[In];
  __shared__ float red[256];
  __shared__ float srt[256];

  if (tid < In) hist[tid] = 0;
  __syncthreads();
  const int4* Lrow4 = (const int4*)(labels + (size_t)b * Nn);
  for (int n = tid; n < Nn / 4; n += 256) {
    const int4 l4 = Lrow4[n];
    atomicAdd(&hist[l4.x], 1);
    atomicAdd(&hist[l4.y], 1);
    atomicAdd(&hist[l4.z], 1);
    atomicAdd(&hist[l4.w], 1);
  }
  __syncthreads();

  // accumulate counts over K-slices
  float cnt[In];
  #pragma unroll
  for (int i = 0; i < In; ++i) cnt[i] = 0.f;
  for (int s = 0; s < KSLICES; ++s) {
    const float* p = partial + (size_t)((s * Bsz + b) * Qn + tid) * In;
    #pragma unroll
    for (int i = 0; i < In; ++i) cnt[i] += p[i];
  }

  float total = 0.f, maxc = -1.f; int mode = 0;
  #pragma unroll
  for (int i = 0; i < In; ++i) {           // ascending scan + strict '>' == first argmax
    total += cnt[i];
    if (cnt[i] > maxc) { maxc = cnt[i]; mode = i; }
  }
  const float inter = maxc;
  const float uni   = total + (float)hist[mode] - inter;
  const float iou   = inter / fmaxf(uni, 1.f);
  const bool  has   = total > 0.f;
  const bool  pos   = has && (iou >= 0.5f);
  const bool  neg   = ((!has) || (iou <= 0.3f)) && !pos;
  const float train = pos ? 1.f : 0.f;

  const float s    = sim_score[b * Qn + tid];
  const float bce  = bce_logits(s, train);
  const float lpos = bce * train;
  const float lneg = bce * (1.f - train);

  const float num_pos  = blockReduce256(train, red, tid);
  const float num_neg  = blockReduce256(neg ? 1.f : 0.f, red, tid);
  const float sum_lpos = blockReduce256(lpos, red, tid);

  // descending bitonic sort of loss_neg (256 elems, 256 threads)
  srt[tid] = lneg;
  __syncthreads();
  for (int k = 2; k <= 256; k <<= 1) {
    for (int j = k >> 1; j > 0; j >>= 1) {
      const int ixj = tid ^ j;
      if (ixj > tid) {
        const float va = srt[tid], vb = srt[ixj];
        const bool up = ((tid & k) == 0);
        if (up ? (va < vb) : (va > vb)) { srt[tid] = vb; srt[ixj] = va; }
      }
      __syncthreads();
    }
  }

  const float n_hard = fminf(num_neg, 3.0f * num_pos);
  const float sel    = ((float)tid < n_hard) ? srt[tid] : 0.f;
  const float sum_sel = blockReduce256(sel, red, tid);

  if (tid == 0) {
    simpart[b * 3 + 0] = num_pos;
    simpart[b * 3 + 1] = sum_lpos;
    simpart[b * 3 + 2] = sum_sel;
  }
}

// -----------------------------------------------------------------------------
// Kernel 3: dice + focal per gathered row (B*G = 64 rows of N elements),
// streamed as float4 (rows are 16B aligned: N*4B = 160000B).
// -----------------------------------------------------------------------------
__global__ __launch_bounds__(256)
void dicefocal_kernel(const float* __restrict__ logits,
                      const float* __restrict__ gt,
                      const int*   __restrict__ pred_inds,
                      float*       __restrict__ rowpart) {
  const int rowid = blockIdx.x;   // 0..63
  const int b  = rowid / Gn;
  const int tid = threadIdx.x;
  const int qi = pred_inds[rowid];                 // [B,G] flat
  const float4* pr4 = (const float4*)(logits + (size_t)(b * Qn + qi) * Nn);
  const float4* tr4 = (const float4*)(gt + (size_t)rowid * Nn);

  float s_pt = 0.f, s_p = 0.f, s_t = 0.f, s_fl = 0.f;
  for (int n = tid; n < Nn / 4; n += 256) {
    const float4 s4 = pr4[n];
    const float4 t4 = tr4[n];
    #pragma unroll
    for (int e = 0; e < 4; ++e) {
      const float sv = (&s4.x)[e], tv = (&t4.x)[e];
      const float p  = 1.f / (1.f + expf(-sv));
      const float ce = bce_logits(sv, tv);
      const float pt = p * tv + (1.f - p) * (1.f - tv);
      const float at = 0.25f * tv + 0.75f * (1.f - tv);
      const float om = 1.f - pt;
      s_pt += p * tv; s_p += p; s_t += tv;
      s_fl += at * ce * om * om;
    }
  }
  __shared__ float red[256];
  s_pt = blockReduce256(s_pt, red, tid);
  s_p  = blockReduce256(s_p,  red, tid);
  s_t  = blockReduce256(s_t,  red, tid);
  s_fl = blockReduce256(s_fl, red, tid);
  if (tid == 0) {
    rowpart[rowid * 2 + 0] = 1.f - (2.f * s_pt + 1.f) / (s_p + s_t + 1.f);
    rowpart[rowid * 2 + 1] = s_fl / (float)Nn;
  }
}

// -----------------------------------------------------------------------------
// Kernel 4: final scalar combine.
// -----------------------------------------------------------------------------
__global__ __launch_bounds__(64)
void final_kernel(const float* __restrict__ simpart,
                  const float* __restrict__ rowpart,
                  float*       __restrict__ out) {
  __shared__ float red[64];
  const int tid = threadIdx.x;   // 64 threads == 64 rows

  red[tid] = rowpart[tid * 2 + 0];
  __syncthreads();
  #pragma unroll
  for (int s = 32; s > 0; s >>= 1) { if (tid < s) red[tid] += red[tid + s]; __syncthreads(); }
  const float dice_sum = red[0];
  __syncthreads();

  red[tid] = rowpart[tid * 2 + 1];
  __syncthreads();
  #pragma unroll
  for (int s = 32; s > 0; s >>= 1) { if (tid < s) red[tid] += red[tid + s]; __syncthreads(); }
  const float focal_sum = red[0];
  __syncthreads();

  if (tid == 0) {
    float tp = 0.f, sall = 0.f;
    #pragma unroll
    for (int b = 0; b < Bsz; ++b) {
      tp   += simpart[b * 3 + 0];
      sall += simpart[b * 3 + 1] + simpart[b * 3 + 2];
    }
    const float sim = (tp > 0.f) ? (sall / fmaxf(tp, 1.f)) : 0.f;
    const float inv = 1.f / ((float)Gn + 1e-6f) / (float)Bsz;
    out[0] = sim + dice_sum * inv + focal_sum * inv;
  }
}

// -----------------------------------------------------------------------------
extern "C" void kernel_launch(void* const* d_in, const int* in_sizes, int n_in,
                              void* d_out, int out_size, void* d_ws, size_t ws_size,
                              hipStream_t stream) {
  (void)in_sizes; (void)n_in; (void)out_size; (void)ws_size;
  const float* mask_logits = (const float*)d_in[0];   // [B,Q,N] f32
  const float* sim_score   = (const float*)d_in[1];   // [B,Q]   f32
  const float* inst_gt     = (const float*)d_in[2];   // [B,G,N] f32
  const int*   labels      = (const int*)  d_in[3];   // [B,N]   i32
  const int*   pred_inds   = (const int*)  d_in[4];   // [B,G]   i32

  float* ws      = (float*)d_ws;
  float* partial = ws;                                        // KSLICES*B*Q*I = 327680 f
  float* simpart = ws + (size_t)KSLICES * Bsz * Qn * In;      // 12 f
  float* rowpart = simpart + 16;                              // 128 f

  counts_wmma_kernel<<<dim3(16, Bsz, KSLICES), 32, 0, stream>>>(mask_logits, labels, partial);
  sim_kernel<<<Bsz, 256, 0, stream>>>(sim_score, labels, partial, simpart);
  dicefocal_kernel<<<Bsz * Gn, 256, 0, stream>>>(mask_logits, inst_gt, pred_inds, rowpart);
  final_kernel<<<1, 64, 0, stream>>>(simpart, rowpart, (float*)d_out);
}
